// SpatialGraphBatch_9594956939716
// MI455X (gfx1250) — matrix operand order repaired and checked by reading, not compiled, not run on hardware
//
#include <hip/hip_runtime.h>
#include <hip/hip_bf16.h>

// ---------------------------------------------------------------------------
// SpatialGraphBatch GCN for MI455X (gfx1250, wave32, WMMA bf16).
// prep_weights: one-time f32->bf16 transpose of W1/W2 into d_ws (Wt[n][k]).
// spatial_gcn_kernel: one workgroup (256 thr = 8 waves) per graph:
//   M (19x19 propagation matrix) from edge weights,
//   H = sigmoid(M @ (X @ W1) + b1),  O = sigmoid(M @ (H @ W2) + b2)
// GEMM B-fragments stream straight from L2 (each weight element is consumed
// by exactly one lane), A-fragments from LDS. All LDS operands are zero-padded
// to 32 rows so fragment gathers are unconditional (no EXEC juggling).
// ---------------------------------------------------------------------------

typedef __attribute__((ext_vector_type(16))) __bf16 v16bf;
typedef __attribute__((ext_vector_type(8)))  float  v8f;

#define AI __attribute__((always_inline))

#define N_BOX   19
#define N_EDGE  171
#define NPAD    32
#define FEAT0   512
#define FEAT1   256
// LDS strides (elements) padded so row-stride in dwords is odd -> no bank conflicts
#define XSTR    514   // X rows (bf16), 32 rows (19..31 zeroed)
#define H1STR   258   // H1 rows (bf16), 32 rows (19..31 zeroed)
#define MSTR    34    // M-matrix rows (bf16), 32 rows
#define SSTR    34    // transposed stage rows (bf16), K slots 19..33 zeroed

struct SharedMem {
  union {
    unsigned short X[NPAD * XSTR];               // 32896 B : features, bf16, row-major
    unsigned short stageT[256 * SSTR];           // 17408 B : (A@W)^T, bf16 [n][node]
  } big;
  unsigned short H1[NPAD * H1STR];               // 16512 B : hidden layer, bf16, row-major
  unsigned short Mmat[NPAD * MSTR];              //  2176 B : padded 32x32 propagation matrix
  float nodeCX[N_BOX], nodeCY[N_BOX], nodeX0[N_BOX], nodeY0[N_BOX];
  float nodeNX[N_BOX], nodeNY[N_BOX], nodeMask[N_BOX];
  float wvals[N_EDGE];
  unsigned char eR[N_EDGE], eC[N_EDGE];
  float dinv[N_BOX];
  float red[256];
};  // ~53 KB total

// float -> bf16 bits, round-to-nearest-even
__device__ AI inline unsigned short f2bf(float f) {
  unsigned int x = __float_as_uint(f);
  unsigned int r = (x + 0x7FFFu + ((x >> 16) & 1u)) >> 16;
  return (unsigned short)r;
}

// A-fragment gather (16x32 bf16) from LDS, row-major source, 32 valid rows:
// lanes 0-15: M=lane,    VGPR j: K=k0+2j
// lanes16-31: M=lane-16, VGPR j: K=k0+16+2j
__device__ AI inline v16bf gatherA(const unsigned short* base, int stride,
                                   int m0, int k0) {
  const int lane = threadIdx.x & 31;
  const int m    = m0 + (lane & 15);
  const int hi   = lane >> 4;
  union { v16bf v; unsigned int u[8]; } r;
  const unsigned int* row = (const unsigned int*)(base + m * stride);
#pragma unroll
  for (int j = 0; j < 8; ++j) {
    r.u[j]     = row[(k0 + 16 * hi + 2 * j) >> 1];
  }
  return r.v;
}

// B-fragment gather (32x16 bf16) from LDS, transposed source (row n holds K).
// lanes 0-15: N=lane, K=0..15 ; lanes16-31: N=lane-16, K=16..31
__device__ AI inline v16bf gatherB(const unsigned short* base, int stride, int n0) {
  const int lane = threadIdx.x & 31;
  const int n    = n0 + (lane & 15);
  const int kb   = (lane >> 4) << 4;
  union { v16bf v; unsigned int u[8]; } r;
  const unsigned int* row = (const unsigned int*)(base + n * stride);
#pragma unroll
  for (int j = 0; j < 8; ++j)
    r.u[j] = row[(kb >> 1) + j];
  return r.v;
}

// B-fragment gather straight from global (pre-transposed bf16 weights, ld = Kdim).
// Per lane the 8 dwords are contiguous in K -> two global_load_b128.
__device__ AI inline v16bf gatherBg(const unsigned short* __restrict__ Wt,
                                    int ld, int n0, int k0) {
  const int lane = threadIdx.x & 31;
  const int n    = n0 + (lane & 15);
  const int kb   = k0 + ((lane >> 4) << 4);
  union { v16bf v; uint4 q[2]; } r;
  const uint4* p = (const uint4*)(Wt + (size_t)n * ld + kb);
  r.q[0] = p[0];
  r.q[1] = p[1];
  return r.v;
}

__device__ AI inline v8f wmma_bf16(v16bf a, v16bf b, v8f c) {
  return __builtin_amdgcn_wmma_f32_16x16x32_bf16(
      /*neg_a=*/false, a, /*neg_b=*/false, b,
      /*c_mod=*/(short)0, c, /*reuse_a=*/false, /*reuse_b=*/false);
}

__device__ AI inline v8f zero8() {
  v8f z = {0.f, 0.f, 0.f, 0.f, 0.f, 0.f, 0.f, 0.f};
  return z;
}

// One GEMM layer over a 256-wide N slice:
//   Z = M @ (A[32xK] @ Wt[nBase:nBase+256][:]) + bias ; sigmoid ; -> H1 (bf16) or out (f32)
__device__ AI inline void run_layer(SharedMem& s,
                                    const unsigned short* Asrc, int aStride, int Kdim,
                                    const unsigned short* __restrict__ Wt, int nBase,
                                    const float* __restrict__ bias,
                                    unsigned short* H1dst, float* outG) {
  const int t     = threadIdx.x;
  const int wv    = t >> 5;        // wave 0..7, owns N columns [32*wv, 32*wv+32)
  const int lane  = t & 31;
  const int hi    = lane >> 4;
  const int nLane = lane & 15;

  v8f acc[2][2];
  acc[0][0] = zero8(); acc[0][1] = zero8();
  acc[1][0] = zero8(); acc[1][1] = zero8();

  const int n0base = nBase + 32 * wv;
  const int nPanels = Kdim / 32;
#pragma unroll 2
  for (int p = 0; p < nPanels; ++p) {
    const int k0 = p * 32;
    const v16bf a0 = gatherA(Asrc, aStride, 0,  k0);
    const v16bf a1 = gatherA(Asrc, aStride, 16, k0);
    const v16bf b0 = gatherBg(Wt, Kdim, n0base,      k0);
    const v16bf b1 = gatherBg(Wt, Kdim, n0base + 16, k0);
    acc[0][0] = wmma_bf16(a0, b0, acc[0][0]);
    acc[0][1] = wmma_bf16(a0, b1, acc[0][1]);
    acc[1][0] = wmma_bf16(a1, b0, acc[1][0]);
    acc[1][1] = wmma_bf16(a1, b1, acc[1][1]);
  }
  __syncthreads();  // all waves done reading Asrc (unions with stageT)

  // Stage (A@W) transposed into LDS as bf16: stageT[n][node]; zero node-pad slots
  {
    unsigned short* rz = s.big.stageT + t * SSTR;  // thread t owns row n=t
    rz[19] = 0;
    unsigned int* rzd = (unsigned int*)(rz + 20);  // K slots 20..33
#pragma unroll
    for (int j = 0; j < 7; ++j) rzd[j] = 0u;
#pragma unroll
    for (int rt = 0; rt < 2; ++rt)
#pragma unroll
      for (int ci = 0; ci < 2; ++ci) {
        const int n = 32 * wv + 16 * ci + nLane;
#pragma unroll
        for (int r = 0; r < 8; ++r) {
          const int m = rt * 16 + r + 8 * hi;
          if (m < N_BOX) s.big.stageT[n * SSTR + m] = f2bf(acc[rt][ci][r]);
        }
      }
  }
  __syncthreads();

  // Aggregate: Z = M(32x32) @ stage(32 x 256) -- one WMMA per 16x16 tile (K=32)
#pragma unroll
  for (int rt = 0; rt < 2; ++rt)
#pragma unroll
    for (int ci = 0; ci < 2; ++ci) {
      const int n0 = 32 * wv + 16 * ci;
      const v16bf a = gatherA(s.Mmat, MSTR, rt * 16, 0);
      const v16bf b = gatherB(s.big.stageT, SSTR, n0);
      v8f z = wmma_bf16(a, b, zero8());
      const int nloc = n0 + nLane;
      const float bv = bias[nBase + nloc];
#pragma unroll
      for (int r = 0; r < 8; ++r) {
        const int m = rt * 16 + r + 8 * hi;
        if (m < N_BOX) {
          const float v = 1.0f / (1.0f + __expf(-(z[r] + bv)));
          if (H1dst) H1dst[m * H1STR + nloc] = f2bf(v);
          else       outG[(size_t)m * FEAT0 + nBase + nloc] = v;
        }
      }
    }
  __syncthreads();
}

// One-time: transpose + convert weights to bf16 into workspace.
// Wt1[n][k] (256 x 512) at ws[0..131071], Wt2[n][k] (512 x 256) after it.
__global__ __launch_bounds__(256) void prep_weights(
    const float* __restrict__ W1, const float* __restrict__ W2,
    unsigned short* __restrict__ ws) {
  const int id = blockIdx.x * 256 + threadIdx.x;  // 0 .. 512*256-1
  unsigned short* Wt1 = ws;
  unsigned short* Wt2 = ws + FEAT1 * FEAT0;
  {
    const int k = id / FEAT1, n = id % FEAT1;     // W1: 512 x 256
    Wt1[(size_t)n * FEAT0 + k] = f2bf(W1[id]);
  }
  {
    const int k = id / FEAT0, n = id % FEAT0;     // W2: 256 x 512
    Wt2[(size_t)n * FEAT1 + k] = f2bf(W2[id]);
  }
}

__global__ __launch_bounds__(256) void spatial_gcn_kernel(
    const float* __restrict__ loc, const float* __restrict__ feat,
    const unsigned char* __restrict__ mask, const float* __restrict__ past,
    const unsigned short* __restrict__ Wt1, const float* __restrict__ b1,
    const unsigned short* __restrict__ Wt2, const float* __restrict__ b2,
    float* __restrict__ out) {
  __shared__ SharedMem s;
  const int g = blockIdx.x;
  const int t = threadIdx.x;

  // ---- Phase 0: X -> LDS (bf16), zero pads, per-node precompute, zero M ----
  const float* Xg = feat + (size_t)g * (N_BOX * FEAT0);
  for (int idx = t; idx < (N_BOX * FEAT0) / 4; idx += 256) {
    const float4 v = reinterpret_cast<const float4*>(Xg)[idx];
    const int f = idx * 4, m = f >> 9, k = f & 511;
    unsigned int* dst = (unsigned int*)(s.big.X + m * XSTR + k);
    dst[0] = (unsigned int)f2bf(v.x) | ((unsigned int)f2bf(v.y) << 16);
    dst[1] = (unsigned int)f2bf(v.z) | ((unsigned int)f2bf(v.w) << 16);
  }
  // zero X rows 19..31 (13 rows x 257 dwords)
  for (int i = t; i < 13 * (XSTR / 2); i += 256) {
    const int row = N_BOX + i / (XSTR / 2), dw = i % (XSTR / 2);
    ((unsigned int*)(s.big.X + row * XSTR))[dw] = 0u;
  }
  // zero H1 rows 19..31 (13 rows x 129 dwords)
  for (int i = t; i < 13 * (H1STR / 2); i += 256) {
    const int row = N_BOX + i / (H1STR / 2), dw = i % (H1STR / 2);
    ((unsigned int*)(s.H1 + row * H1STR))[dw] = 0u;
  }
  if (t < N_BOX) {
    const float* L = loc + (size_t)g * (N_BOX * 5) + t * 5;
    const float cx = 0.5f * (L[1] + L[3]);
    const float cy = 0.5f * (L[2] + L[4]);
    const float* P0 = past + (size_t)g * (5 * N_BOX * 6) + t * 6;                    // t = 0
    const float* P4 = past + (size_t)g * (5 * N_BOX * 6) + 4 * (N_BOX * 6) + t * 6;  // t = T-1
    const float x0 = (P0[3] - P0[1]) * 1280.f, y0 = (P0[4] - P0[2]) * 720.f;
    const float x4 = (P4[3] - P4[1]) * 1280.f, y4 = (P4[4] - P4[2]) * 720.f;
    const float dx = x4 - x0, dy = y4 - y0;
    float len = sqrtf(dx * dx + dy * dy);
    len = (len == 0.f) ? 1.f : len;
    s.nodeCX[t] = cx; s.nodeCY[t] = cy;
    s.nodeX0[t] = x0; s.nodeY0[t] = y0;
    s.nodeNX[t] = dx / len; s.nodeNY[t] = dy / len;
    s.nodeMask[t] = mask[(size_t)g * N_BOX + t] ? 1.f : 0.f;
  }
  for (int i = t; i < NPAD * MSTR; i += 256) s.Mmat[i] = 0;
  __syncthreads();

  // ---- Phase 1: edge weights ----
  if (t < N_EDGE) {
    int e = t, r = 0;
    while (e >= 18 - r) { e -= 18 - r; ++r; }
    const int c = r + 1 + e;
    s.eR[t] = (unsigned char)r; s.eC[t] = (unsigned char)c;
    const float dcx = s.nodeCX[r] - s.nodeCX[c], dcy = s.nodeCY[r] - s.nodeCY[c];
    const float d = dcx * dcx + dcy * dcy;
    const float ax = s.nodeX0[r] + s.nodeNX[r] - s.nodeX0[c] - s.nodeNX[c];
    const float ay = s.nodeY0[r] + s.nodeNY[r] - s.nodeY0[c] - s.nodeNY[c];
    const float d1 = sqrtf(ax * ax + ay * ay);
    const float bx = s.nodeX0[r] - s.nodeX0[c], by = s.nodeY0[r] - s.nodeY0[c];
    const float d2 = sqrtf(bx * bx + by * by);
    float w = __expf(-(0.7f * d + 0.3f * (d1 - d2)));
    w = (s.nodeMask[r] != 0.f && s.nodeMask[c] != 0.f) ? w : 0.f;
    s.wvals[t] = w;
  }
  __syncthreads();

  // ---- Phase 2: min/max normalization ----
  s.red[t] = (t < N_EDGE) ? s.wvals[t] : 3.402823466e38f;
  __syncthreads();
  for (int sh = 128; sh > 0; sh >>= 1) {
    if (t < sh) s.red[t] = fminf(s.red[t], s.red[t + sh]);
    __syncthreads();
  }
  const float wmin = s.red[0];
  __syncthreads();
  s.red[t] = (t < N_EDGE) ? s.wvals[t] : -3.402823466e38f;
  __syncthreads();
  for (int sh = 128; sh > 0; sh >>= 1) {
    if (t < sh) s.red[t] = fmaxf(s.red[t], s.red[t + sh]);
    __syncthreads();
  }
  const float wmax = s.red[0];
  __syncthreads();
  if (t < N_EDGE) s.wvals[t] = (s.wvals[t] - wmin) / (wmax - wmin);
  __syncthreads();

  // ---- Phase 3: degrees & normalized propagation matrix M ----
  if (t < N_BOX) {
    float deg = 1.f;  // self loop
    for (int e = 0; e < N_EDGE; ++e)
      if (s.eC[e] == t) deg += s.wvals[e];
    s.dinv[t] = 1.0f / sqrtf(deg);
  }
  __syncthreads();
  if (t < N_EDGE) {
    const int r = s.eR[t], c = s.eC[t];
    s.Mmat[c * MSTR + r] = f2bf(s.dinv[r] * s.wvals[t] * s.dinv[c]);
  } else if (t < N_EDGE + N_BOX) {
    const int i = t - N_EDGE;
    s.Mmat[i * MSTR + i] = f2bf(s.dinv[i] * s.dinv[i]);
  }
  __syncthreads();

  // ---- Phase 4: two GCN layers (layer 2 split into two 256-wide N halves) ----
  float* outG = out + (size_t)g * (N_BOX * FEAT0);
  run_layer(s, s.big.X, XSTR, FEAT0, Wt1, 0,   b1, s.H1, nullptr);
  run_layer(s, s.H1,  H1STR, FEAT1, Wt2, 0,   b2, nullptr, outG);
  run_layer(s, s.H1,  H1STR, FEAT1, Wt2, 256, b2, nullptr, outG);
}

extern "C" void kernel_launch(void* const* d_in, const int* in_sizes, int n_in,
                              void* d_out, int out_size, void* d_ws, size_t ws_size,
                              hipStream_t stream) {
  const float*         loc  = (const float*)d_in[0];
  const float*         feat = (const float*)d_in[1];
  const unsigned char* msk  = (const unsigned char*)d_in[2];  // jnp bool -> 1 byte
  const float*         past = (const float*)d_in[3];
  const float*         W1   = (const float*)d_in[4];
  const float*         b1   = (const float*)d_in[5];
  const float*         W2   = (const float*)d_in[6];
  const float*         b2   = (const float*)d_in[7];
  float* out = (float*)d_out;
  unsigned short* ws = (unsigned short*)d_ws;   // needs 2*512*256*2 = 512 KB

  const int B = in_sizes[0] / (N_BOX * 5);

  hipLaunchKernelGGL(prep_weights, dim3((FEAT0 * FEAT1) / 256), dim3(256), 0, stream,
                     W1, W2, ws);
  hipLaunchKernelGGL(spatial_gcn_kernel, dim3(B), dim3(256), 0, stream,
                     loc, feat, msk, past,
                     ws, b1, ws + FEAT1 * FEAT0, b2, out);
}
